// Net_LSTM_59150289600672
// MI455X (gfx1250) — matrix operand order repaired and checked by reading, not compiled, not run on hardware
//
#include <hip/hip_runtime.h>
#include <hip/hip_bf16.h>
#include <math.h>

#define BATCH 16

typedef __attribute__((ext_vector_type(8)))  _Float16 v8h;
typedef __attribute__((ext_vector_type(16))) _Float16 v16h;
typedef __attribute__((ext_vector_type(8)))  float    v8f;

#define BN_INV 0.99999500003749937f   // 1/sqrt(1+1e-5)
#define NEG_MAX (-3.402823466e38f)

// ---------------- WMMA fragment helpers (CDNA5 16x16x32 f16 layouts, ISA 7.12.2) -------------

// A (16x32 f16): lane holds row m = lane&15; halves h<8 -> k = h + 8*(lane>>4),
//                halves h>=8 -> k = 16 + (h-8) + 8*(lane>>4). Two contiguous 8-half chunks.
__device__ __forceinline__ v16h load_a_frag(const _Float16* act, int row0, int ldk, int k0, int lane) {
  const _Float16* p = act + (row0 + (lane & 15)) * ldk + k0 + ((lane >> 4) << 3);
  v8h lo = *(const v8h*)p;
  v8h hi = *(const v8h*)(p + 16);
  return __builtin_shufflevector(lo, hi, 0,1,2,3,4,5,6,7,8,9,10,11,12,13,14,15);
}

// B (32x16 f16) pre-packed per tile: packed[((nt*ktiles+kt)*32 + lane)*16 + h]
__device__ __forceinline__ v16h load_b_frag(const _Float16* wp, int nt, int kt, int ktiles, int lane) {
  return *(const v16h*)(wp + (((nt * ktiles + kt) * 32 + lane) << 4));
}

__device__ __forceinline__ v8f wmma_f16(v16h a, v16h b, v8f c) {
  return __builtin_amdgcn_wmma_f32_16x16x32_f16(false, a, false, b, (short)0, c, false, false);
}

// D tile (16x16 f32): lane holds col n = lane&15, rows m = v + 8*(lane>>4).
// Apply bias + ReLU + eval-BN, store fp16 row-major.
__device__ __forceinline__ void store_tile_bnrelu(v8f acc, _Float16* out, int row0, int ld, int nbase,
                                                  const float* bias, const float* gam, const float* bet,
                                                  int lane) {
  int n = nbase + (lane & 15);
  float bi = bias[n], ga = gam[n], be = bet[n];
  _Float16* p = out + (row0 + ((lane >> 4) << 3)) * ld + n;
  #pragma unroll
  for (int v = 0; v < 8; ++v) {
    float x = acc[v] + bi;
    x = x > 0.f ? x : 0.f;
    x = ga * (x * (float)BN_INV) + be;
    p[v * ld] = (_Float16)x;
  }
}

// ---------------- wave-level layers -----------------------------------------------------------

template<int KPAD, int NOUT>
__device__ void layer_rows64(const _Float16* in, _Float16* out, const _Float16* wp,
                             const float* bias, const float* gam, const float* bet, int lane) {
  constexpr int KT = KPAD / 32;
  #pragma unroll 1
  for (int mt = 0; mt < 4; ++mt)
    #pragma unroll 1
    for (int nt = 0; nt < NOUT / 16; ++nt) {
      v8f acc = {};
      #pragma unroll
      for (int kt = 0; kt < KT; ++kt)
        acc = wmma_f16(load_a_frag(in, mt * 16, KPAD, kt * 32, lane),
                       load_b_frag(wp, nt, kt, KT, lane), acc);
      store_tile_bnrelu(acc, out, mt * 16, NOUT, nt * 16, bias, gam, bet, lane);
    }
}

template<int KPAD, int NOUT>
__device__ void layer_rows64_max(const _Float16* in, const char* flags, const _Float16* wp,
                                 const float* bias, const float* gam, const float* bet,
                                 _Float16* xout_row, int lane) {
  constexpr int KT = KPAD / 32;
  #pragma unroll 1
  for (int nt = 0; nt < NOUT / 16; ++nt) {
    int n = nt * 16 + (lane & 15);
    float bi = bias[n], ga = gam[n], be = bet[n];
    float vmax = NEG_MAX;
    #pragma unroll 1
    for (int mt = 0; mt < 4; ++mt) {
      v8f acc = {};
      #pragma unroll
      for (int kt = 0; kt < KT; ++kt)
        acc = wmma_f16(load_a_frag(in, mt * 16, KPAD, kt * 32, lane),
                       load_b_frag(wp, nt, kt, KT, lane), acc);
      int mb = mt * 16 + ((lane >> 4) << 3);
      #pragma unroll
      for (int v = 0; v < 8; ++v) {
        float x = acc[v] + bi;
        x = x > 0.f ? x : 0.f;
        x = ga * (x * (float)BN_INV) + be;
        if (flags[mb + v]) vmax = fmaxf(vmax, x);
      }
    }
    vmax = fmaxf(vmax, __shfl_xor(vmax, 16, 32));
    if (lane < 16) xout_row[n] = (_Float16)vmax;
  }
}

template<int KPAD, int NOUT>
__device__ void layer_rows16(const _Float16* in, _Float16* out, const _Float16* wp,
                             const float* bias, const float* gam, const float* bet, int lane) {
  constexpr int KT = KPAD / 32;
  #pragma unroll 1
  for (int nt = 0; nt < NOUT / 16; ++nt) {
    v8f acc = {};
    #pragma unroll
    for (int kt = 0; kt < KT; ++kt)
      acc = wmma_f16(load_a_frag(in, 0, KPAD, kt * 32, lane),
                     load_b_frag(wp, nt, kt, KT, lane), acc);
    store_tile_bnrelu(acc, out, 0, NOUT, nt * 16, bias, gam, bet, lane);
  }
}

// ---------------- weight repack: W(out,in) f32 -> B-fragment-packed fp16 ----------------------

__global__ void repack_w_kernel(const float* __restrict__ W, _Float16* __restrict__ dst,
                                int in_dim, int Kpad, int Nout) {
  int t = blockIdx.x * blockDim.x + threadIdx.x;
  int total = Nout * Kpad;
  if (t >= total) return;
  int ktiles = Kpad >> 5;
  int h    = t & 15;
  int lane = (t >> 4) & 31;
  int tile = t >> 9;
  int kt = tile % ktiles;
  int nt = tile / ktiles;
  int k = kt * 32 + ((lane >> 4) << 3) + (h & 7) + ((h >> 3) << 4);
  int n = nt * 16 + (lane & 15);
  float v = (k < in_dim) ? W[(size_t)n * in_dim + k] : 0.f;
  dst[t] = (_Float16)v;
}

// ---------------- farthest point sampling -----------------------------------------------------

template<int N, int S>
__global__ void fps_kernel(const float* __restrict__ pos, int* __restrict__ idx) {
  __shared__ float px[N], py[N], pz[N], mind[N];
  __shared__ float rv[256];
  __shared__ int   ri[256];
  int b = blockIdx.x, tid = threadIdx.x;
  const float* p = pos + (size_t)b * N * 3;
  for (int i = tid; i < N; i += 256) {
    px[i] = p[i * 3]; py[i] = p[i * 3 + 1]; pz[i] = p[i * 3 + 2];
    mind[i] = 3.402823466e38f;
  }
  __syncthreads();
  int last = 0;
  for (int s = 0; s < S; ++s) {
    if (tid == 0) idx[b * S + s] = last;
    float lx = px[last], ly = py[last], lz = pz[last];
    float bv = -1.f; int bi = 0;
    for (int i = tid; i < N; i += 256) {
      float dx = px[i] - lx, dy = py[i] - ly, dz = pz[i] - lz;
      float d = dx * dx + dy * dy + dz * dz;
      float m = mind[i]; m = d < m ? d : m; mind[i] = m;
      if (m > bv) { bv = m; bi = i; }           // ascending i keeps first max
    }
    rv[tid] = bv; ri[tid] = bi;
    __syncthreads();
    for (int off = 128; off > 0; off >>= 1) {
      if (tid < off) {
        float v2 = rv[tid + off]; int i2 = ri[tid + off];
        if (v2 > rv[tid] || (v2 == rv[tid] && i2 < ri[tid])) { rv[tid] = v2; ri[tid] = i2; }
      }
      __syncthreads();
    }
    last = ri[0];
    __syncthreads();
  }
}

__global__ void gather_posq_kernel(const float* __restrict__ pos, const int* __restrict__ idx,
                                   float* __restrict__ posq, int S, int N) {
  int t = blockIdx.x * blockDim.x + threadIdx.x;
  if (t >= BATCH * S) return;
  int bg = t / S;
  int j = idx[t];
  const float* p = pos + (size_t)(bg * N + j) * 3;
  float* o = posq + (size_t)t * 3;
  o[0] = p[0]; o[1] = p[1]; o[2] = p[2];
}

// first K_NBR in-radius neighbors in index order; -1 padding
__global__ void radius_kernel(const float* __restrict__ pos, const float* __restrict__ posq,
                              int* __restrict__ nbr, int S, int N, float r2) {
  int q = blockIdx.x * blockDim.x + threadIdx.x;
  if (q >= BATCH * S) return;
  int bg = q / S;
  const float* p = pos + (size_t)bg * N * 3;
  float qx = posq[q * 3], qy = posq[q * 3 + 1], qz = posq[q * 3 + 2];
  int* out = nbr + q * 64;
  int cnt = 0;
  for (int j = 0; j < N && cnt < 64; ++j) {
    float dx = p[j * 3] - qx, dy = p[j * 3 + 1] - qy, dz = p[j * 3 + 2] - qz;
    if (dx * dx + dy * dy + dz * dz <= r2) out[cnt++] = j;
  }
  for (; cnt < 64; ++cnt) out[cnt] = -1;
}

// ---------------- fused PointConv: one wave per query, 64 neighbor rows -----------------------

template<int FINX, int FINPAD, int H1, int H2, int FOUT, int NSRC, int WAVES>
__global__ void pointconv_kernel(const float* __restrict__ pos, const _Float16* __restrict__ xsrc,
                                 const float* __restrict__ posq, const int* __restrict__ nbr,
                                 const _Float16* w1, const float* b1, const float* g1, const float* be1,
                                 const _Float16* w2, const float* b2, const float* g2, const float* be2,
                                 const _Float16* w3, const float* b3, const float* g3, const float* be3,
                                 _Float16* __restrict__ xout, int S) {
  extern __shared__ char smem[];
  constexpr int BUFA = 64 * (FINPAD > H2 ? FINPAD : H2);
  constexpr int BUFB = 64 * H1;
  int wave = threadIdx.x >> 5;
  int lane = threadIdx.x & 31;
  _Float16* bufA = (_Float16*)smem + wave * (BUFA + BUFB);
  _Float16* bufB = bufA + BUFA;
  char* flags = smem + (size_t)WAVES * (BUFA + BUFB) * 2 + wave * 64;

  int q = blockIdx.x * WAVES + wave;
  int b = q / S;
  const int* nb = nbr + q * 64;
  const float* pq = posq + (size_t)q * 3;
  float qx = pq[0], qy = pq[1], qz = pq[2];

  // gather: row = [x_j (FINX), pos_j - pos_q (3), zero pad]
  #pragma unroll
  for (int rr = 0; rr < 2; ++rr) {
    int r = lane + rr * 32;
    int j = nb[r];
    flags[r] = (char)(j >= 0);
    int js = j >= 0 ? j : 0;
    _Float16* row = bufA + r * FINPAD;
    int c = 0;
    if (FINX > 0) {
      const _Float16* xs = xsrc + (size_t)(b * NSRC + js) * FINX;
      for (; c < FINX; ++c) row[c] = xs[c];
    }
    const float* ps = pos + (size_t)(b * NSRC + js) * 3;
    row[c] = (_Float16)(ps[0] - qx); ++c;
    row[c] = (_Float16)(ps[1] - qy); ++c;
    row[c] = (_Float16)(ps[2] - qz); ++c;
    for (; c < FINPAD; ++c) row[c] = (_Float16)0.f;
  }
  // per-wave private buffers: same-wave LDS ops are in order, no barrier needed
  layer_rows64<FINPAD, H1>(bufA, bufB, w1, b1, g1, be1, lane);
  layer_rows64<H1, H2>(bufB, bufA, w2, b2, g2, be2, lane);
  layer_rows64_max<H2, FOUT>(bufA, flags, w3, b3, g3, be3, xout + (size_t)q * FOUT, lane);
}

// ---------------- global SA: one block/graph, one 16-row tile per wave ------------------------

__global__ void global_sa_kernel(const _Float16* __restrict__ x2, const float* __restrict__ posq2,
                                 const _Float16* w1, const float* b1, const float* g1, const float* be1,
                                 const _Float16* w2, const float* b2, const float* g2, const float* be2,
                                 const _Float16* w3, const float* b3, const float* g3, const float* be3,
                                 float* __restrict__ gout) {
  extern __shared__ char smem[];
  constexpr int F0 = 288, F1 = 256, F2 = 512;
  int wave = threadIdx.x >> 5, lane = threadIdx.x & 31;
  int b = blockIdx.x;
  _Float16* a0 = (_Float16*)smem + wave * (16 * F0);
  _Float16* a1 = (_Float16*)smem + 8 * 16 * F0 + wave * (16 * F1);
  _Float16* a2 = (_Float16*)smem + 8 * 16 * (F0 + F1) + wave * (16 * F2);
  float* sred = (float*)((_Float16*)smem + 8 * 16 * (F0 + F1 + F2));

  for (int t = lane; t < 16 * F0; t += 32) {
    int r = t / F0, c = t % F0;
    int qr = b * 128 + wave * 16 + r;
    float v;
    if (c < 256)       v = (float)x2[(size_t)qr * 256 + c];
    else if (c < 259)  v = posq2[qr * 3 + (c - 256)];
    else               v = 0.f;
    a0[t] = (_Float16)v;
  }
  layer_rows16<F0, F1>(a0, a1, w1, b1, g1, be1, lane);
  layer_rows16<F1, F2>(a1, a2, w2, b2, g2, be2, lane);

  #pragma unroll 1
  for (int nt = 0; nt < 64; ++nt) {
    v8f acc = {};
    #pragma unroll
    for (int kt = 0; kt < F2 / 32; ++kt)
      acc = wmma_f16(load_a_frag(a2, 0, F2, kt * 32, lane),
                     load_b_frag(w3, nt, kt, F2 / 32, lane), acc);
    int n = nt * 16 + (lane & 15);
    float bi = b3[n], ga = g3[n], be = be3[n];
    float vmax = NEG_MAX;
    #pragma unroll
    for (int v = 0; v < 8; ++v) {
      float x = acc[v] + bi;
      x = x > 0.f ? x : 0.f;
      x = ga * (x * (float)BN_INV) + be;
      vmax = fmaxf(vmax, x);
    }
    vmax = fmaxf(vmax, __shfl_xor(vmax, 16, 32));
    if (lane < 16) sred[wave * 16 + lane] = vmax;
    __syncthreads();
    if (wave == 0 && lane < 16) {
      float m = sred[lane];
      #pragma unroll
      for (int w = 1; w < 8; ++w) m = fmaxf(m, sred[w * 16 + lane]);
      gout[b * 1024 + n] = m;
    }
    __syncthreads();
  }
}

// ---------------- LSTM head (H=1) + linears ---------------------------------------------------

__device__ __forceinline__ float sigm(float z) { return 1.f / (1.f + expf(-z)); }

__global__ void lstm_kernel(const float* __restrict__ g,
                            const float* Wih0, const float* Whh0, const float* bih0, const float* bhh0,
                            const float* Wih1, const float* Whh1, const float* bih1, const float* bhh1,
                            float* __restrict__ h1seq, float* __restrict__ h2seq) {
  int lane = threadIdx.x;
  if (lane >= BATCH) return;
  float wi[4], wh[4], bb[4];
  #pragma unroll
  for (int k = 0; k < 4; ++k) { wi[k] = Wih0[k]; wh[k] = Whh0[k]; bb[k] = bih0[k] + bhh0[k]; }
  float h = 0.f, c = 0.f;
  for (int t = 0; t < 1024; ++t) {
    float x = g[lane * 1024 + t];
    float zi = wi[0] * x + wh[0] * h + bb[0];
    float zf = wi[1] * x + wh[1] * h + bb[1];
    float zg = wi[2] * x + wh[2] * h + bb[2];
    float zo = wi[3] * x + wh[3] * h + bb[3];
    c = sigm(zf) * c + sigm(zi) * tanhf(zg);
    h = sigm(zo) * tanhf(c);
    h1seq[lane * 1024 + t] = h;
  }
  #pragma unroll
  for (int k = 0; k < 4; ++k) { wi[k] = Wih1[k]; wh[k] = Whh1[k]; bb[k] = bih1[k] + bhh1[k]; }
  h = 0.f; c = 0.f;
  for (int t = 0; t < 1024; ++t) {
    float x = h1seq[lane * 1024 + t];
    float zi = wi[0] * x + wh[0] * h + bb[0];
    float zf = wi[1] * x + wh[1] * h + bb[1];
    float zg = wi[2] * x + wh[2] * h + bb[2];
    float zo = wi[3] * x + wh[3] * h + bb[3];
    c = sigm(zf) * c + sigm(zi) * tanhf(zg);
    h = sigm(zo) * tanhf(c);
    h2seq[lane * 1024 + t] = h;
  }
}

__global__ void linear_kernel(const float* __restrict__ in, const float* __restrict__ W,
                              const float* __restrict__ bias, float* __restrict__ out,
                              int Bn, int IN, int OUT) {
  int o = blockIdx.x * blockDim.x + threadIdx.x;
  if (o >= Bn * OUT) return;
  int bi = o / OUT, oc = o % OUT;
  const float* x = in + (size_t)bi * IN;
  const float* w = W + (size_t)oc * IN;
  float acc = bias[oc];
  for (int k = 0; k < IN; ++k) acc += x[k] * w[k];
  out[o] = acc;
}

// ---------------- host orchestration ----------------------------------------------------------

extern "C" void kernel_launch(void* const* d_in, const int* in_sizes, int n_in,
                              void* d_out, int out_size, void* d_ws, size_t ws_size,
                              hipStream_t stream) {
  (void)in_sizes; (void)n_in; (void)out_size; (void)ws_size;
  const float* pos = (const float*)d_in[0];

  char* ws = (char*)d_ws;
  size_t off = 0;
  auto alloc = [&](size_t bytes) -> char* {
    char* p = ws + off;
    off += (bytes + 255) & ~(size_t)255;
    return p;
  };

  int*      idx1  = (int*)alloc((size_t)BATCH * 512 * 4);
  float*    posq1 = (float*)alloc((size_t)BATCH * 512 * 3 * 4);
  int*      nbr1  = (int*)alloc((size_t)BATCH * 512 * 64 * 4);
  _Float16* x1    = (_Float16*)alloc((size_t)BATCH * 512 * 128 * 2);
  int*      idx2  = (int*)alloc((size_t)BATCH * 128 * 4);
  float*    posq2 = (float*)alloc((size_t)BATCH * 128 * 3 * 4);
  int*      nbr2  = (int*)alloc((size_t)BATCH * 128 * 64 * 4);
  _Float16* x2    = (_Float16*)alloc((size_t)BATCH * 128 * 256 * 2);
  float*    gbuf  = (float*)alloc((size_t)BATCH * 1024 * 4);
  float*    h1seq = (float*)alloc((size_t)BATCH * 1024 * 4);
  float*    h2seq = (float*)alloc((size_t)BATCH * 1024 * 4);
  float*    l1out = (float*)alloc((size_t)BATCH * 512 * 4);

  // repack all 9 MLP weight matrices into WMMA B-fragment order (fp16)
  struct LayerDesc { int widx, in, kpad, nout; };
  const LayerDesc L[9] = {
    {2, 3, 32, 64},   {6, 64, 64, 64},   {10, 64, 64, 128},
    {14, 131, 160, 128}, {18, 128, 128, 128}, {22, 128, 128, 256},
    {26, 259, 288, 256}, {30, 256, 256, 512}, {34, 512, 512, 1024},
  };
  _Float16* wp[9];
  for (int i = 0; i < 9; ++i) {
    int total = L[i].nout * L[i].kpad;
    wp[i] = (_Float16*)alloc((size_t)total * 2);
    repack_w_kernel<<<(total + 255) / 256, 256, 0, stream>>>(
        (const float*)d_in[L[i].widx], wp[i], L[i].in, L[i].kpad, L[i].nout);
  }
  auto P = [&](int i) { return (const float*)d_in[i]; };

  // --- SA1: FPS 1024->512, radius 0.2, PointConv 3->64->64->128 ---
  fps_kernel<1024, 512><<<BATCH, 256, 0, stream>>>(pos, idx1);
  gather_posq_kernel<<<(BATCH * 512 + 255) / 256, 256, 0, stream>>>(pos, idx1, posq1, 512, 1024);
  radius_kernel<<<(BATCH * 512 + 255) / 256, 256, 0, stream>>>(pos, posq1, nbr1, 512, 1024,
                                                               (float)(0.2 * 0.2));
  {
    constexpr int WAVES = 4;
    size_t shmem = (size_t)WAVES * (64 * 64 + 64 * 64) * 2 + WAVES * 64;   // 65792 B
    pointconv_kernel<0, 32, 64, 64, 128, 1024, WAVES>
        <<<BATCH * 512 / WAVES, WAVES * 32, shmem, stream>>>(
            pos, (const _Float16*)nullptr, posq1, nbr1,
            wp[0], P(3), P(4), P(5), wp[1], P(7), P(8), P(9), wp[2], P(11), P(12), P(13),
            x1, 512);
  }

  // --- SA2: FPS 512->128, radius 0.4, PointConv 131->128->128->256 ---
  fps_kernel<512, 128><<<BATCH, 256, 0, stream>>>(posq1, idx2);
  gather_posq_kernel<<<(BATCH * 128 + 255) / 256, 256, 0, stream>>>(posq1, idx2, posq2, 128, 512);
  radius_kernel<<<(BATCH * 128 + 255) / 256, 256, 0, stream>>>(posq1, posq2, nbr2, 128, 512,
                                                               (float)(0.4 * 0.4));
  {
    constexpr int WAVES = 2;
    size_t shmem = (size_t)WAVES * (64 * 160 + 64 * 128) * 2 + WAVES * 64; // 73856 B
    pointconv_kernel<128, 160, 128, 128, 256, 512, WAVES>
        <<<BATCH * 128 / WAVES, WAVES * 32, shmem, stream>>>(
            posq1, x1, posq2, nbr2,
            wp[3], P(15), P(16), P(17), wp[4], P(19), P(20), P(21), wp[5], P(23), P(24), P(25),
            x2, 128);
  }

  // --- Global SA: 259(->288)->256->512->1024 + max pool per graph ---
  {
    size_t shmem = (size_t)8 * 16 * (288 + 256 + 512) * 2 + 8 * 16 * 4;    // 270848 B (<=320KB WGP LDS)
    global_sa_kernel<<<BATCH, 256, shmem, stream>>>(
        x2, posq2,
        wp[6], P(27), P(28), P(29), wp[7], P(31), P(32), P(33), wp[8], P(35), P(36), P(37),
        gbuf);
  }

  // --- LSTM head (2 layers, H=1) + linears ---
  lstm_kernel<<<1, 32, 0, stream>>>(gbuf, P(38), P(39), P(40), P(41), P(42), P(43), P(44), P(45),
                                    h1seq, h2seq);
  linear_kernel<<<(BATCH * 512 + 255) / 256, 256, 0, stream>>>(h2seq, P(46), P(47), l1out,
                                                               BATCH, 1024, 512);
  linear_kernel<<<(BATCH * 40 + 255) / 256, 256, 0, stream>>>(l1out, P(48), P(49), (float*)d_out,
                                                              BATCH, 512, 40);
}
